// LatentReasoningBlock_67216238182707
// MI455X (gfx1250) — matrix operand order repaired and checked by reading, not compiled, not run on hardware
//
#include <hip/hip_runtime.h>
#include <math.h>
#include <stdint.h>

#define Dd   1024
#define DIi  4096
#define Bb   4
#define Ll   4096
#define Rr   (Bb * Ll)     // 16384 rows
#define Mm   8             // thought memory slots

typedef __bf16 bfx16 __attribute__((ext_vector_type(16)));
typedef float  fx8   __attribute__((ext_vector_type(8)));
typedef int    v4i   __attribute__((ext_vector_type(4)));

__device__ __forceinline__ unsigned short f2bf(float f) {
  unsigned u = __float_as_uint(f);
  unsigned r = 0x7FFFu + ((u >> 16) & 1u);   // round-to-nearest-even
  return (unsigned short)((u + r) >> 16);
}

// ---- async global->LDS copy (CDNA5), with sync fallback ----
#if defined(__HIP_DEVICE_COMPILE__) && __has_builtin(__builtin_amdgcn_global_load_async_to_lds_b128)
#define HAVE_ASYNC_LDS 1
typedef __attribute__((address_space(1))) v4i* as1_v4i;
typedef __attribute__((address_space(3))) v4i* as3_v4i;
__device__ __forceinline__ void async_b128(const void* g, void* l) {
  as1_v4i gp = (as1_v4i)(uintptr_t)g;                 // generic -> global AS
  as3_v4i lp = (as3_v4i)(unsigned int)(uintptr_t)l;   // generic -> LDS offset (low 32b)
  __builtin_amdgcn_global_load_async_to_lds_b128(gp, lp, 0, 0);
}
#else
#define HAVE_ASYNC_LDS 0
#endif

template<int N>
__device__ __forceinline__ void wait_async() {
#if defined(__HIP_DEVICE_COMPILE__)
#if __has_builtin(__builtin_amdgcn_s_wait_asynccnt)
  __builtin_amdgcn_s_wait_asynccnt(N);
#else
  if (N == 0) asm volatile("s_wait_asynccnt 0x0" ::: "memory");
  else        asm volatile("s_wait_asynccnt 0x3" ::: "memory");
#endif
#endif
}

__device__ __forceinline__ float blockSum(float v, float* red, int tid) {
  red[tid] = v; __syncthreads();
  for (int o = 128; o > 0; o >>= 1) { if (tid < o) red[tid] += red[tid + o]; __syncthreads(); }
  float r = red[0]; __syncthreads();
  return r;
}
__device__ __forceinline__ float blockMax(float v, float* red, int tid) {
  red[tid] = v; __syncthreads();
  for (int o = 128; o > 0; o >>= 1) { if (tid < o) red[tid] = fmaxf(red[tid], red[tid + o]); __syncthreads(); }
  float r = red[0]; __syncthreads();
  return r;
}

// -------- weight fp32 [K][N] -> bf16 transposed [N][K] --------
__global__ void k_cvt_t(const float* __restrict__ s, unsigned short* __restrict__ d,
                        int K, int N) {
  size_t total = (size_t)K * N;
  size_t i = (size_t)blockIdx.x * blockDim.x + threadIdx.x;
  size_t stride = (size_t)gridDim.x * blockDim.x;
  for (; i < total; i += stride) {
    int n = (int)(i / K), k = (int)(i - (size_t)n * K);
    d[i] = f2bf(s[(size_t)k * N + n]);
  }
}

// -------- per-row: LayerNorm (-> bf16) + entropy gate --------
__global__ __launch_bounds__(256)
void k_ln_ent(const float* __restrict__ hidden, const float* __restrict__ g1,
              const float* __restrict__ b1, unsigned short* __restrict__ lnbf,
              float* __restrict__ gate, float inv_logd) {
  __shared__ float red[256];
  int row = blockIdx.x, tid = threadIdx.x;
  const float* xr = hidden + (size_t)row * Dd;
  float v[4], s = 0.f, mx = -3.4e38f;
#pragma unroll
  for (int j = 0; j < 4; ++j) { v[j] = xr[tid + 256 * j]; s += v[j]; mx = fmaxf(mx, v[j]); }
  float mean = blockSum(s, red, tid) * (1.f / Dd);
  float vs = 0.f;
#pragma unroll
  for (int j = 0; j < 4; ++j) { float d0 = v[j] - mean; vs += d0 * d0; }
  float var = blockSum(vs, red, tid) * (1.f / Dd);
  float rstd = rsqrtf(var + 1e-6f);
  float rmax = blockMax(mx, red, tid);
  float s1 = 0.f, s2 = 0.f;
#pragma unroll
  for (int j = 0; j < 4; ++j) { float e = expf(v[j] - rmax); s1 += e; s2 += v[j] * e; }
  float S1 = blockSum(s1, red, tid);
  float S2 = blockSum(s2, red, tid);
  if (tid == 0) {
    float ent = (rmax + logf(S1) - S2 / S1) * inv_logd;
    gate[row] = (ent > 0.5f) ? 1.f : 0.f;
  }
#pragma unroll
  for (int j = 0; j < 4; ++j) {
    int i = tid + 256 * j;
    lnbf[(size_t)row * Dd + i] = f2bf((v[j] - mean) * rstd * g1[i] + b1[i]);
  }
}

// -------- bf16 WMMA GEMM: C = A[MxK] * Bt[NxK]^T (+bias, epilogue) --------
// Block tile 128(M) x 64(N); 8 waves along M, each wave -> 16x64 (4 accumulators).
// Double-buffered async global->LDS staging (overlap DMA with WMMA).
// EPI 0: out_bf16 = bf16(gelu(acc + bias))       (FFN up-proj)
// EPI 1: out_f32  = acc + bias + res             (down-proj / attn, residual)
template<int EPI>
__global__ __launch_bounds__(256)
void k_gemm(const unsigned short* __restrict__ A,    // [M][K] bf16
            const unsigned short* __restrict__ Bt,   // [N][K] bf16 (pre-transposed)
            const float* __restrict__ bias, const float* __restrict__ res,
            float* __restrict__ outF, unsigned short* __restrict__ outB,
            int M, int N, int K) {
  __shared__ unsigned short As[2][128][40];   // 128x32 per buf, row stride 80B
  __shared__ unsigned short BsT[2][64][40];   // 64(N) x 32(K) per buf
  int tid = threadIdx.x;
  int bm = blockIdx.y * 128, bn = blockIdx.x * 64;
  int lane = tid & 31, wv = tid >> 5;
  int wm = wv * 16;
  int am = lane & 15, kb = (lane >> 4) * 8;      // A: lanes 16-31 carry K+8 / K+24
  int bnc = lane & 15, kb2 = (lane >> 4) * 8;    // B: lanes 16-31 carry K+8 / K+24
  fx8 c[4];
#pragma unroll
  for (int s4 = 0; s4 < 4; ++s4) c[s4] = (fx8){0.f,0.f,0.f,0.f,0.f,0.f,0.f,0.f};
  int ar = tid >> 1, ac = (tid & 1) * 16;        // A stage: 2x b128 per thread
  int brn = tid >> 2, bkc = (tid & 3) * 8;       // B stage: 1x b128 per thread

  auto issue = [&](int buf, int k0) {
    const unsigned short* ga0 = A  + (size_t)(bm + ar)  * K + k0 + ac;
    const unsigned short* gb0 = Bt + (size_t)(bn + brn) * K + k0 + bkc;
#if HAVE_ASYNC_LDS
    async_b128(ga0,     &As[buf][ar][ac]);
    async_b128(ga0 + 8, &As[buf][ar][ac + 8]);
    async_b128(gb0,     &BsT[buf][brn][bkc]);
#else
    *(uint4*)&As[buf][ar][ac]     = *(const uint4*)ga0;
    *(uint4*)&As[buf][ar][ac + 8] = *(const uint4*)(ga0 + 8);
    *(uint4*)&BsT[buf][brn][bkc]  = *(const uint4*)gb0;
#endif
  };

  issue(0, 0);
  for (int k0 = 0; k0 < K; k0 += 32) {
    int cur = (k0 >> 5) & 1;
    bool more = (k0 + 32) < K;
    if (more) issue(cur ^ 1, k0 + 32);           // prefetch next tile (other buffer)
#if HAVE_ASYNC_LDS
    if (more) wait_async<3>(); else wait_async<0>();   // current tile's 3 copies done
#endif
    __syncthreads();
    union { uint4 q[2]; bfx16 v; } af;
    af.q[0] = *(const uint4*)&As[cur][wm + am][kb];        // K = kb..kb+7
    af.q[1] = *(const uint4*)&As[cur][wm + am][16 + kb];   // K = 16+kb..16+kb+7
#pragma unroll
    for (int s4 = 0; s4 < 4; ++s4) {
      union { uint4 q[2]; bfx16 v; } bf;
      bf.q[0] = *(const uint4*)&BsT[cur][s4 * 16 + bnc][kb2];      // K = kb2..kb2+7
      bf.q[1] = *(const uint4*)&BsT[cur][s4 * 16 + bnc][16 + kb2]; // K = 16+kb2..16+kb2+7
      c[s4] = __builtin_amdgcn_wmma_f32_16x16x32_bf16(false, af.v, false, bf.v,
                                                      (short)0, c[s4], false, false);
    }
    __syncthreads();   // all reads of 'cur' done before it is overwritten next round
  }
  int crow = bm + wm + (lane >> 4) * 8;
#pragma unroll
  for (int s4 = 0; s4 < 4; ++s4) {
    int col = bn + s4 * 16 + (lane & 15);
    float bv = bias[col];
#pragma unroll
    for (int i = 0; i < 8; ++i) {
      int r = crow + i;
      float v0 = c[s4][i] + bv;
      if constexpr (EPI == 0) {
        v0 = 0.5f * v0 * (1.f + erff(v0 * 0.70710678118654752f));
        outB[(size_t)r * N + col] = f2bf(v0);
      } else {
        outF[(size_t)r * N + col] = v0 + res[(size_t)r * N + col];
      }
    }
  }
}

// -------- scores = h1 @ W_comp + b_comp  [R,8] --------
__global__ __launch_bounds__(256)
void k_scores(const float* __restrict__ h1, const float* __restrict__ Wc,
              const float* __restrict__ bc, float* __restrict__ scores) {
  __shared__ float rowb[Dd];
  __shared__ float red[256];
  int row = blockIdx.x, tid = threadIdx.x, lane = tid & 31, wv = tid >> 5;
#pragma unroll
  for (int j = 0; j < 4; ++j) rowb[tid + 256 * j] = h1[(size_t)row * Dd + tid + 256 * j];
  __syncthreads();
  float p = 0.f;
  for (int i = lane; i < Dd; i += 32) p += rowb[i] * Wc[i * Mm + wv];
  red[tid] = p; __syncthreads();
  for (int o = 16; o > 0; o >>= 1) { if (lane < o) red[tid] += red[tid + o]; __syncthreads(); }
  if (lane == 0) scores[(size_t)row * Mm + wv] = red[tid] + bc[wv];
}

// -------- softmax-over-L column stats --------
__global__ __launch_bounds__(256)
void k_colstats(const float* __restrict__ scores, float* __restrict__ cmax, float* __restrict__ csum) {
  __shared__ float red[256];
  int b = blockIdx.x >> 3, m = blockIdx.x & 7, tid = threadIdx.x;
  float mx = -3.4e38f;
  for (int l = tid; l < Ll; l += 256) mx = fmaxf(mx, scores[((size_t)b * Ll + l) * Mm + m]);
  float M0 = blockMax(mx, red, tid);
  float s = 0.f;
  for (int l = tid; l < Ll; l += 256) s += expf(scores[((size_t)b * Ll + l) * Mm + m] - M0);
  float S = blockSum(s, red, tid);
  if (tid == 0) { cmax[blockIdx.x] = M0; csum[blockIdx.x] = S; }
}

// -------- memory[b,m,d] partials: sum_l exp(s-max) * h1 --------
__global__ __launch_bounds__(256)
void k_mem_partial(const float* __restrict__ scores, const float* __restrict__ h1,
                   const float* __restrict__ cmax, float* __restrict__ part) {
  __shared__ float sc[32][8];
  int tid = threadIdx.x;
  int d = blockIdx.x * 256 + tid;
  int seg = blockIdx.y, b = blockIdx.z;
  float cm[8];
#pragma unroll
  for (int m = 0; m < 8; ++m) cm[m] = cmax[b * 8 + m];
  float av[8] = {0.f,0.f,0.f,0.f,0.f,0.f,0.f,0.f};
  for (int l0 = seg * 256; l0 < seg * 256 + 256; l0 += 32) {
    sc[tid >> 3][tid & 7] = expf(scores[((size_t)(b * Ll + l0 + (tid >> 3))) * Mm + (tid & 7)] - cm[tid & 7]);
    __syncthreads();
    for (int j = 0; j < 32; ++j) {
      float h = h1[((size_t)(b * Ll + l0 + j)) * Dd + d];
#pragma unroll
      for (int m = 0; m < 8; ++m) av[m] += sc[j][m] * h;
    }
    __syncthreads();
  }
#pragma unroll
  for (int m = 0; m < 8; ++m)
    part[(((size_t)(b * 16 + seg)) * 8 + m) * Dd + d] = av[m];
}

__global__ __launch_bounds__(256)
void k_mem_reduce(const float* __restrict__ part, const float* __restrict__ csum,
                  float* __restrict__ mem) {
  int idx = blockIdx.x * 256 + threadIdx.x;      // [b(4)][m(8)][d(1024)]
  int b = idx >> 13, m = (idx >> 10) & 7, d = idx & 1023;
  float s = 0.f;
  for (int seg = 0; seg < 16; ++seg) s += part[(((size_t)(b * 16 + seg)) * 8 + m) * Dd + d];
  mem[idx] = s / csum[b * 8 + m];
}

// -------- ctx = softmax_m(scores) @ memory  -> bf16 --------
__global__ __launch_bounds__(256)
void k_ctx(const float* __restrict__ scores, const float* __restrict__ mem,
           unsigned short* __restrict__ ctxbf) {
  int row = blockIdx.x, tid = threadIdx.x;
  int b = row >> 12;
  float s[8], mx = -3.4e38f, se = 0.f;
#pragma unroll
  for (int m = 0; m < 8; ++m) { s[m] = scores[(size_t)row * Mm + m]; mx = fmaxf(mx, s[m]); }
#pragma unroll
  for (int m = 0; m < 8; ++m) { s[m] = expf(s[m] - mx); se += s[m]; }
  float inv = 1.f / se;
  for (int d = tid; d < Dd; d += 256) {
    float c = 0.f;
#pragma unroll
    for (int m = 0; m < 8; ++m) c += s[m] * mem[((size_t)(b * 8 + m)) * Dd + d];
    ctxbf[(size_t)row * Dd + d] = f2bf(c * inv);
  }
}

// -------- entropy gate + ACT halting; writes hidden (=h_new) + acc/cum --------
__global__ __launch_bounds__(256)
void k_gate_halt(const float* __restrict__ h2, float* __restrict__ hidden,
                 const float* __restrict__ gate, const float* __restrict__ Wh,
                 const float* __restrict__ bh, float* __restrict__ cum,
                 float* __restrict__ acc) {
  __shared__ float red[256];
  int row = blockIdx.x, tid = threadIdx.x;
  float g = gate[row];
  float hn[4], p = 0.f;
#pragma unroll
  for (int j = 0; j < 4; ++j) {
    int i = tid + 256 * j;
    size_t off = (size_t)row * Dd + i;
    hn[j] = (g > 0.5f) ? h2[off] : hidden[off];
    p += hn[j] * Wh[i];
  }
  float dot = blockSum(p, red, tid);
  float pr = 1.f / (1.f + expf(-(dot + bh[0])));
  float c = cum[row];
  float w = pr * (1.f - c);
#pragma unroll
  for (int j = 0; j < 4; ++j) {
    size_t off = (size_t)row * Dd + tid + 256 * j;
    acc[off] += w * hn[j];
    hidden[off] = hn[j];
  }
  if (tid == 0) cum[row] = c + w;
}

// -------- out = LN(acc + (1-cum)*hidden) --------
__global__ __launch_bounds__(256)
void k_final(const float* __restrict__ acc, const float* __restrict__ hidden,
             const float* __restrict__ cum, const float* __restrict__ g2,
             const float* __restrict__ b2, float* __restrict__ out) {
  __shared__ float red[256];
  int row = blockIdx.x, tid = threadIdx.x;
  float c = 1.f - cum[row];
  float v[4], s = 0.f;
#pragma unroll
  for (int j = 0; j < 4; ++j) {
    size_t off = (size_t)row * Dd + tid + 256 * j;
    v[j] = acc[off] + c * hidden[off];
    s += v[j];
  }
  float mean = blockSum(s, red, tid) * (1.f / Dd);
  float vs = 0.f;
#pragma unroll
  for (int j = 0; j < 4; ++j) { float d0 = v[j] - mean; vs += d0 * d0; }
  float var = blockSum(vs, red, tid) * (1.f / Dd);
  float rstd = rsqrtf(var + 1e-6f);
#pragma unroll
  for (int j = 0; j < 4; ++j) {
    int i = tid + 256 * j;
    out[(size_t)row * Dd + i] = (v[j] - mean) * rstd * g2[i] + b2[i];
  }
}

extern "C" void kernel_launch(void* const* d_in, const int* in_sizes, int n_in,
                              void* d_out, int out_size, void* d_ws, size_t ws_size,
                              hipStream_t stream) {
  const float* x      = (const float*)d_in[0];
  const float* g1     = (const float*)d_in[1];
  const float* b1     = (const float*)d_in[2];
  const float* W_up   = (const float*)d_in[3];
  const float* b_up   = (const float*)d_in[4];
  const float* W_down = (const float*)d_in[5];
  const float* b_down = (const float*)d_in[6];
  const float* W_halt = (const float*)d_in[7];
  const float* b_halt = (const float*)d_in[8];
  const float* W_comp = (const float*)d_in[9];
  const float* b_comp = (const float*)d_in[10];
  const float* W_attn = (const float*)d_in[11];
  const float* b_attn = (const float*)d_in[12];
  const float* g2     = (const float*)d_in[13];
  const float* b2     = (const float*)d_in[14];

  char* p = (char*)d_ws;
  auto alloc = [&](size_t bytes) -> void* {
    void* r = (void*)p;
    p += (bytes + 255) & ~(size_t)255;
    return r;
  };
  float*          hidden = (float*)alloc((size_t)Rr * Dd * 4);
  float*          acc    = (float*)alloc((size_t)Rr * Dd * 4);
  float*          h1     = (float*)alloc((size_t)Rr * Dd * 4);
  unsigned short* Ubf    = (unsigned short*)alloc((size_t)Rr * DIi * 2);
  unsigned short* lnctx  = (unsigned short*)alloc((size_t)Rr * Dd * 2);  // LN out, later ctx
  unsigned short* WupT   = (unsigned short*)alloc((size_t)Dd * DIi * 2); // [DI][D]
  unsigned short* WdownT = (unsigned short*)alloc((size_t)DIi * Dd * 2); // [D][DI]
  unsigned short* WattnT = (unsigned short*)alloc((size_t)Dd * Dd * 2);  // [D][D]
  float*          cum    = (float*)alloc((size_t)Rr * 4);
  float*          gate   = (float*)alloc((size_t)Rr * 4);
  float*          scores = (float*)alloc((size_t)Rr * Mm * 4);
  float*          cmax   = (float*)alloc(32 * 4);
  float*          csum   = (float*)alloc(32 * 4);
  float*          part   = (float*)alloc((size_t)Bb * 16 * Mm * Dd * 4);
  float*          mem    = (float*)alloc((size_t)Bb * Mm * Dd * 4);

  (void)hipMemcpyAsync(hidden, x, (size_t)Rr * Dd * 4, hipMemcpyDeviceToDevice, stream);
  (void)hipMemsetAsync(acc, 0, (size_t)Rr * Dd * 4, stream);
  (void)hipMemsetAsync(cum, 0, (size_t)Rr * 4, stream);

  // weights: bf16, transposed to [N][K] so GEMM B-tiles are K-contiguous
  k_cvt_t<<<2048, 256, 0, stream>>>(W_up,   WupT,   Dd,  DIi);
  k_cvt_t<<<2048, 256, 0, stream>>>(W_down, WdownT, DIi, Dd);
  k_cvt_t<<<1024, 256, 0, stream>>>(W_attn, WattnT, Dd,  Dd);

  const float inv_logd = 1.0f / logf((float)Dd);

  for (int step = 0; step < 4; ++step) {
    k_ln_ent<<<Rr, 256, 0, stream>>>(hidden, g1, b1, lnctx, gate, inv_logd);
    // U = gelu(LN(h) @ W_up + b_up)  -> bf16
    k_gemm<0><<<dim3(DIi / 64, Rr / 128), 256, 0, stream>>>(
        lnctx, WupT, b_up, nullptr, nullptr, Ubf, Rr, DIi, Dd);
    // h1 = hidden + U @ W_down + b_down
    k_gemm<1><<<dim3(Dd / 64, Rr / 128), 256, 0, stream>>>(
        Ubf, WdownT, b_down, hidden, h1, nullptr, Rr, Dd, DIi);
    // thought memory
    k_scores<<<Rr, 256, 0, stream>>>(h1, W_comp, b_comp, scores);
    k_colstats<<<32, 256, 0, stream>>>(scores, cmax, csum);
    k_mem_partial<<<dim3(4, 16, 4), 256, 0, stream>>>(scores, h1, cmax, part);
    k_mem_reduce<<<(Bb * Mm * Dd) / 256, 256, 0, stream>>>(part, csum, mem);
    k_ctx<<<Rr, 256, 0, stream>>>(scores, mem, lnctx);
    // h2 = h1 + ctx @ W_attn + b_attn   (in-place residual: each elem RMW once)
    k_gemm<1><<<dim3(Dd / 64, Rr / 128), 256, 0, stream>>>(
        lnctx, WattnT, b_attn, h1, h1, nullptr, Rr, Dd, Dd);
    // gate + ACT halting; updates hidden/acc/cum
    k_gate_halt<<<Rr, 256, 0, stream>>>(h1, hidden, gate, W_halt, b_halt, cum, acc);
  }
  k_final<<<Rr, 256, 0, stream>>>(acc, hidden, cum, g2, b2, (float*)d_out);
}
